// ReactiveGRU_51187420234363
// MI455X (gfx1250) — matrix operand order repaired and checked by reading, not compiled, not run on hardware
//
#include <hip/hip_runtime.h>
#include <math.h>

#define PP 16384
#define HH 256
#define AA 128
#define EE 16
#define DD 1024
#define NPOUT 64
#define NGOUT 32
#define EPSV 1e-5f

typedef __bf16 v16bf __attribute__((ext_vector_type(16)));
typedef __bf16 v8bf  __attribute__((ext_vector_type(8)));
typedef __bf16 v2bf  __attribute__((ext_vector_type(2)));
typedef float  v8f   __attribute__((ext_vector_type(8)));

__device__ __forceinline__ float bn_apply(float x, const float* g, const float* b,
                                          const float* m, const float* v, int c) {
  return (x - m[c]) * rsqrtf(v[c] + EPSV) * g[c] + b[c];
}

// ---- WMMA tile GEMM helpers. Layouts per CDNA5 ISA §7.12.2:
//   A lane t: row m=t&15; elems 0..7 = K kb..kb+7, 8..15 = K kb+16..kb+23, kb=(t<16)?0:8
//   B lane t: col n=t&15; same K order as A (elems 0..7 = kb..kb+7, 8..15 = kb+16..kb+23)

__device__ __forceinline__ v16bf load_a_frag(const __bf16* ar, int k0) {
  v8bf alo = *(const v8bf*)(ar + k0);        // ds_load_b128
  v8bf ahi = *(const v8bf*)(ar + k0 + 16);   // ds_load_b128
  return __builtin_shufflevector(alo, ahi, 0, 1, 2, 3, 4, 5, 6, 7,
                                 8, 9, 10, 11, 12, 13, 14, 15);
}

// single N-tile
__device__ __forceinline__ v8f gemm_k(const __bf16* __restrict__ A, int lda,
                                      const __bf16* __restrict__ W, int ldw,
                                      int K, int n0, int lane, v8f acc) {
  const __bf16* ar = A + (lane & 15) * lda + ((lane & 16) ? 8 : 0);
  const __bf16* wr = W + (size_t)(n0 + (lane & 15)) * ldw + ((lane & 16) ? 8 : 0);
  for (int k0 = 0; k0 < K; k0 += 32) {
    v16bf a = load_a_frag(ar, k0);
    v16bf b = load_a_frag(wr, k0);      // 2x global_load_b128
    acc = __builtin_amdgcn_wmma_f32_16x16x32_bf16(false, a, false, b, (short)0, acc,
                                                  false, false);
  }
  return acc;
}

// dual N-tile: one A fragment feeds two WMMAs (halves LDS reads)
__device__ __forceinline__ void gemm_k2(const __bf16* __restrict__ A, int lda,
                                        const __bf16* __restrict__ W, int ldw,
                                        int K, int n0a, int n0b, int lane,
                                        v8f& acc0, v8f& acc1) {
  const __bf16* ar  = A + (lane & 15) * lda + ((lane & 16) ? 8 : 0);
  int wofs = ((lane & 16) ? 8 : 0);
  const __bf16* wr0 = W + (size_t)(n0a + (lane & 15)) * ldw + wofs;
  const __bf16* wr1 = W + (size_t)(n0b + (lane & 15)) * ldw + wofs;
  for (int k0 = 0; k0 < K; k0 += 32) {
    v16bf a  = load_a_frag(ar, k0);
    v16bf b0 = load_a_frag(wr0, k0);
    v16bf b1 = load_a_frag(wr1, k0);
    acc0 = __builtin_amdgcn_wmma_f32_16x16x32_bf16(false, a, false, b0, (short)0, acc0,
                                                   false, false);
    acc1 = __builtin_amdgcn_wmma_f32_16x16x32_bf16(false, a, false, b1, (short)0, acc1,
                                                   false, false);
  }
}

// ---- prep: f32 -> bf16 weight conversion ----
__global__ void k_cvt(const float* __restrict__ src, __bf16* __restrict__ dst, int n) {
  int i = blockIdx.x * 256 + threadIdx.x;
  if (i < n) dst[i] = (__bf16)src[i];
}

// Ws (A=128,H=256) -> WsTB (H=256,A=128) bf16
__global__ void k_transcvt(const float* __restrict__ Ws, __bf16* __restrict__ WsT) {
  int i = blockIdx.x * 256 + threadIdx.x;
  int a = i >> 8, h = i & 255;
  WsT[h * AA + a] = (__bf16)Ws[i];
}

// ---- attn[p,e] = ( spn_r[p,e,:] . (Ws^T te_p) + bs.te_p ) * temperature ----
__global__ void k_attn(const float* __restrict__ temp, const float* __restrict__ spat,
                       const float* __restrict__ hg, const float* __restrict__ hb,
                       const float* __restrict__ hm, const float* __restrict__ hv,
                       const __bf16* __restrict__ WtB, const float* __restrict__ bt,
                       const __bf16* __restrict__ WsTB, const float* __restrict__ bs,
                       float* __restrict__ attnT) {
  __shared__ __align__(16) __bf16 Ash[16 * HH];
  __shared__ float  teF[16 * AA];
  __shared__ __align__(16) __bf16 teB[16 * AA];
  __shared__ float  uF[16 * HH];
  __shared__ float  cF[16];
  int tid = threadIdx.x, lane = tid & 31, wave = tid >> 5;
  int p0 = blockIdx.x * 16;

  for (int i = tid; i < 16 * HH / 2; i += 256) {  // BN(temp) tile, packed stores
    int m = i >> 7, kp = (i & 127) * 2;
    const float* r = temp + (size_t)(p0 + m) * HH + kp;
    v2bf w;
    w[0] = (__bf16)bn_apply(r[0], hg, hb, hm, hv, kp);
    w[1] = (__bf16)bn_apply(r[1], hg, hb, hm, hv, kp + 1);
    *(v2bf*)(Ash + m * HH + kp) = w;
  }
  __syncthreads();

  { // te = BN(temp) @ Wt^T + bt   (M=16, K=256, N=128; one tile per wave)
    int n0 = wave * 16;
    v8f acc = {};
    acc = gemm_k(Ash, HH, WtB, HH, HH, n0, lane, acc);
    int n = n0 + (lane & 15), mb = (lane & 16) ? 8 : 0;
#pragma unroll
    for (int j = 0; j < 8; ++j) {
      float v = acc[j] + bt[n];
      teF[(mb + j) * AA + n] = v;
      teB[(mb + j) * AA + n] = (__bf16)v;
    }
  }
  __syncthreads();

  if (tid < 16) { // c_p = bs . te_p
    float c = 0.f;
    for (int a = 0; a < AA; ++a) c += bs[a] * teF[tid * AA + a];
    cF[tid] = c;
  }
  { // u = te @ Ws  (M=16, K=128, N=256); two tiles per wave, shared A frags
    int n0a = wave * 16, n0b = (wave + 8) * 16;
    v8f acc0 = {}, acc1 = {};
    gemm_k2(teB, AA, WsTB, AA, AA, n0a, n0b, lane, acc0, acc1);
    int nl = lane & 15, mb = (lane & 16) ? 8 : 0;
#pragma unroll
    for (int j = 0; j < 8; ++j) {
      uF[(mb + j) * HH + n0a + nl] = acc0[j];
      uF[(mb + j) * HH + n0b + nl] = acc1[j];
    }
  }
  __syncthreads();

  { // attn logits, stored transposed (E,P) for the softmax reduction
    int p = tid & 15, e = tid >> 4;
    const float* sp = spat + (size_t)(p0 + p) * (HH * EE) + e * HH;
    float d = 0.f;
    for (int h = 0; h < HH; ++h) {
      int ch = e * 16 + (h >> 4);
      d += bn_apply(sp[h], hg, hb, hm, hv, ch) * uF[p * HH + h];
    }
    attnT[(size_t)e * PP + p0 + p] = (d + cF[p]) * 1.4142135623730951f; // E/sqrt(A)
  }
}

// softmax over persons (dim 0); one block per column e, contiguous row reads
__global__ void k_softmax(const float* __restrict__ attnT, float* __restrict__ smax) {
  __shared__ float red[256];
  int e = blockIdx.x, tid = threadIdx.x;
  const float* row = attnT + (size_t)e * PP;
  float m = -INFINITY;
  for (int p = tid; p < PP; p += 256) m = fmaxf(m, row[p]);
  red[tid] = m; __syncthreads();
  for (int s = 128; s > 0; s >>= 1) {
    if (tid < s) red[tid] = fmaxf(red[tid], red[tid + s]);
    __syncthreads();
  }
  float mx = red[0]; __syncthreads();
  float sum = 0.f;
  for (int p = tid; p < PP; p += 256) sum += __expf(row[p] - mx);
  red[tid] = sum; __syncthreads();
  for (int s = 128; s > 0; s >>= 1) {
    if (tid < s) red[tid] += red[tid + s];
    __syncthreads();
  }
  if (tid == 0) { smax[e] = mx; smax[16 + e] = red[0]; }
}

// Hagg[p,h] = sum_e BN(spn_r[p,e,h]) * w[p,e]
__global__ void k_hagg(const float* __restrict__ spat,
                       const float* __restrict__ hg, const float* __restrict__ hb,
                       const float* __restrict__ hm, const float* __restrict__ hv,
                       const float* __restrict__ attnT, const float* __restrict__ smax,
                       float* __restrict__ Hagg) {
  __shared__ float wsm[EE];
  int p = blockIdx.x, h = threadIdx.x;
  if (h < EE) wsm[h] = __expf(attnT[(size_t)h * PP + p] - smax[h]) / smax[16 + h];
  __syncthreads();
  float acc = 0.f;
#pragma unroll
  for (int e = 0; e < EE; ++e) {
    int f = e * HH + h;
    acc += bn_apply(spat[(size_t)p * (HH * EE) + f], hg, hb, hm, hv, f >> 4) * wsm[e];
  }
  Hagg[(size_t)p * HH + h] = acc;
}

// feature_e = relu(BN(feature) @ Wf^T) -> catB[:, 0:256]  (bf16)
__global__ void k_feat(const float* __restrict__ feature,
                       const float* __restrict__ fg, const float* __restrict__ fb,
                       const float* __restrict__ fm, const float* __restrict__ fv,
                       const __bf16* __restrict__ WfB, __bf16* __restrict__ catB) {
  __shared__ __align__(16) __bf16 Ash[16 * DD];
  int tid = threadIdx.x, lane = tid & 31, wave = tid >> 5;
  int p0 = blockIdx.x * 16;
  for (int i = tid; i < 16 * DD / 2; i += 256) {
    int m = i >> 9, kp = (i & 511) * 2;
    const float* r = feature + (size_t)(p0 + m) * DD + kp;
    v2bf w;
    w[0] = (__bf16)bn_apply(r[0], fg, fb, fm, fv, kp);
    w[1] = (__bf16)bn_apply(r[1], fg, fb, fm, fv, kp + 1);
    *(v2bf*)(Ash + m * DD + kp) = w;
  }
  __syncthreads();
  {
    int n0a = wave * 16, n0b = (wave + 8) * 16;
    v8f acc0 = {}, acc1 = {};
    gemm_k2(Ash, DD, WfB, DD, DD, n0a, n0b, lane, acc0, acc1);
    int nl = lane & 15, mb = (lane & 16) ? 8 : 0;
#pragma unroll
    for (int j = 0; j < 8; ++j) {
      catB[(size_t)(p0 + mb + j) * 512 + n0a + nl] = (__bf16)fmaxf(acc0[j], 0.f);
      catB[(size_t)(p0 + mb + j) * 512 + n0b + nl] = (__bf16)fmaxf(acc1[j], 0.f);
    }
  }
}

// H_e = relu(BN2([temp,Hagg]) @ Wh2^T) -> catB[:, 256:512]  (bf16)
__global__ void k_he(const float* __restrict__ temp, const float* __restrict__ Hagg,
                     const float* __restrict__ g2, const float* __restrict__ b2,
                     const float* __restrict__ m2, const float* __restrict__ v2,
                     const __bf16* __restrict__ Wh2B, __bf16* __restrict__ catB) {
  __shared__ __align__(16) __bf16 Ash[16 * 512];
  int tid = threadIdx.x, lane = tid & 31, wave = tid >> 5;
  int p0 = blockIdx.x * 16;
  for (int i = tid; i < 16 * 512 / 2; i += 256) {
    int m = i >> 8, kp = (i & 255) * 2;
    const float* r = (kp < HH) ? temp + (size_t)(p0 + m) * HH + kp
                               : Hagg + (size_t)(p0 + m) * HH + (kp - HH);
    v2bf w;
    w[0] = (__bf16)bn_apply(r[0], g2, b2, m2, v2, kp);
    w[1] = (__bf16)bn_apply(r[1], g2, b2, m2, v2, kp + 1);
    *(v2bf*)(Ash + m * 512 + kp) = w;
  }
  __syncthreads();
  {
    int n0a = wave * 16, n0b = (wave + 8) * 16;
    v8f acc0 = {}, acc1 = {};
    gemm_k2(Ash, 512, Wh2B, 512, 512, n0a, n0b, lane, acc0, acc1);
    int nl = lane & 15, mb = (lane & 16) ? 8 : 0;
#pragma unroll
    for (int j = 0; j < 8; ++j) {
      catB[(size_t)(p0 + mb + j) * 512 + HH + n0a + nl] = (__bf16)fmaxf(acc0[j], 0.f);
      catB[(size_t)(p0 + mb + j) * 512 + HH + n0b + nl] = (__bf16)fmaxf(acc1[j], 0.f);
    }
  }
}

// ReLuGRU cell: z,r sigmoid gates, n relu gate, hidden = (1-z)*n + z*prev
__global__ void k_gru(const float* __restrict__ prev, const __bf16* __restrict__ catB,
                      const __bf16* __restrict__ wihB, const __bf16* __restrict__ whhB,
                      const float* __restrict__ gbias, float* __restrict__ hidden_out) {
  __shared__ __align__(16) __bf16 Acat[16 * 512];
  __shared__ __align__(16) __bf16 Aprev[16 * HH];
  __shared__ __align__(16) __bf16 Apr[16 * HH];
  __shared__ float zb[16 * HH];
  __shared__ float rb[16 * HH];
  __shared__ float nb[16 * HH];
  int tid = threadIdx.x, lane = tid & 31, wave = tid >> 5;
  int p0 = blockIdx.x * 16;
  { // cat tile is already bf16: bulk uint4 copy (16 rows x 512 = 16 KB)
    const uint4* s = (const uint4*)(catB + (size_t)p0 * 512);
    uint4* d = (uint4*)Acat;
    for (int i = tid; i < 16 * 512 * 2 / 16; i += 256) d[i] = s[i];
  }
  for (int i = tid; i < 16 * HH / 2; i += 256) {
    int m = i >> 7, kp = (i & 127) * 2;
    const float* r = prev + (size_t)(p0 + m) * HH + kp;
    v2bf w; w[0] = (__bf16)r[0]; w[1] = (__bf16)r[1];
    *(v2bf*)(Aprev + m * HH + kp) = w;
  }
  __syncthreads();
  for (int g = 0; g < 2; ++g) {  // z (g=0), r (g=1)
    const __bf16* Wi = wihB + (size_t)g * HH * 512;
    const __bf16* Wh = whhB + (size_t)g * HH * HH;
    const float* bg = gbias + g * HH;
    float* outb = (g == 0) ? zb : rb;
    int n0a = wave * 16, n0b = (wave + 8) * 16;
    v8f acc0 = {}, acc1 = {};
    gemm_k2(Acat, 512, Wi, 512, 512, n0a, n0b, lane, acc0, acc1);
    gemm_k2(Aprev, HH, Wh, HH, HH, n0a, n0b, lane, acc0, acc1);
    int nl = lane & 15, mb = (lane & 16) ? 8 : 0;
#pragma unroll
    for (int j = 0; j < 8; ++j) {
      float v0 = acc0[j] + bg[n0a + nl];
      float v1 = acc1[j] + bg[n0b + nl];
      outb[(mb + j) * HH + n0a + nl] = 1.f / (1.f + __expf(-v0));
      outb[(mb + j) * HH + n0b + nl] = 1.f / (1.f + __expf(-v1));
    }
  }
  __syncthreads();
  for (int i = tid; i < 16 * HH / 2; i += 256) {  // prev * r as bf16 A tile
    int m = i >> 7, kp = (i & 127) * 2;
    const float* r = prev + (size_t)(p0 + m) * HH + kp;
    v2bf w;
    w[0] = (__bf16)(r[0] * rb[m * HH + kp]);
    w[1] = (__bf16)(r[1] * rb[m * HH + kp + 1]);
    *(v2bf*)(Apr + m * HH + kp) = w;
  }
  __syncthreads();
  { // n gate
    const __bf16* Wi = wihB + (size_t)2 * HH * 512;
    const __bf16* Wh = whhB + (size_t)2 * HH * HH;
    const float* bg = gbias + 2 * HH;
    int n0a = wave * 16, n0b = (wave + 8) * 16;
    v8f acc0 = {}, acc1 = {};
    gemm_k2(Acat, 512, Wi, 512, 512, n0a, n0b, lane, acc0, acc1);
    gemm_k2(Apr, HH, Wh, HH, HH, n0a, n0b, lane, acc0, acc1);
    int nl = lane & 15, mb = (lane & 16) ? 8 : 0;
#pragma unroll
    for (int j = 0; j < 8; ++j) {
      nb[(mb + j) * HH + n0a + nl] = fmaxf(acc0[j] + bg[n0a + nl], 0.f);
      nb[(mb + j) * HH + n0b + nl] = fmaxf(acc1[j] + bg[n0b + nl], 0.f);
    }
  }
  __syncthreads();
  for (int i = tid; i < 16 * HH; i += 256) {
    int m = i >> 8, k = i & 255;
    float z = zb[i], pv = prev[(size_t)(p0 + m) * HH + k];
    hidden_out[(size_t)(p0 + m) * HH + k] = (1.f - z) * nb[i] + z * pv;
  }
}

// heads: hn = BN1(hidden); person = hn@Wp^T+bp, group = hn@Wg^T+bg
// 96 threads = 3 waves; wave w -> tiles 2w, 2w+1 (0..3 Wp, 4..5 Wg)
__global__ void k_heads(const float* __restrict__ hidden,
                        const float* __restrict__ g1, const float* __restrict__ b1,
                        const float* __restrict__ m1, const float* __restrict__ v1,
                        const __bf16* __restrict__ WpB, const float* __restrict__ bp,
                        const __bf16* __restrict__ WgB, const float* __restrict__ bgv,
                        float* __restrict__ pout, float* __restrict__ gout) {
  __shared__ __align__(16) __bf16 Ash[16 * HH];
  int tid = threadIdx.x, lane = tid & 31, wave = tid >> 5;
  int p0 = blockIdx.x * 16;
  for (int i = tid; i < 16 * HH / 2; i += 96) {
    int m = i >> 7, kp = (i & 127) * 2;
    const float* r = hidden + (size_t)(p0 + m) * HH + kp;
    v2bf w;
    w[0] = (__bf16)bn_apply(r[0], g1, b1, m1, v1, kp);
    w[1] = (__bf16)bn_apply(r[1], g1, b1, m1, v1, kp + 1);
    *(v2bf*)(Ash + m * HH + kp) = w;
  }
  __syncthreads();
  if (wave < 2) {  // tiles 2w, 2w+1 of Wp
    int n0a = (2 * wave) * 16, n0b = (2 * wave + 1) * 16;
    v8f acc0 = {}, acc1 = {};
    gemm_k2(Ash, HH, WpB, HH, HH, n0a, n0b, lane, acc0, acc1);
    int nl = lane & 15, mb = (lane & 16) ? 8 : 0;
#pragma unroll
    for (int j = 0; j < 8; ++j) {
      pout[(size_t)(p0 + mb + j) * NPOUT + n0a + nl] = acc0[j] + bp[n0a + nl];
      pout[(size_t)(p0 + mb + j) * NPOUT + n0b + nl] = acc1[j] + bp[n0b + nl];
    }
  } else {         // tiles 0,1 of Wg
    int n0a = 0, n0b = 16;
    v8f acc0 = {}, acc1 = {};
    gemm_k2(Ash, HH, WgB, HH, HH, n0a, n0b, lane, acc0, acc1);
    int nl = lane & 15, mb = (lane & 16) ? 8 : 0;
#pragma unroll
    for (int j = 0; j < 8; ++j) {
      gout[(size_t)(p0 + mb + j) * NGOUT + n0a + nl] = acc0[j] + bgv[n0a + nl];
      gout[(size_t)(p0 + mb + j) * NGOUT + n0b + nl] = acc1[j] + bgv[n0b + nl];
    }
  }
}

extern "C" void kernel_launch(void* const* d_in, const int* in_sizes, int n_in,
                              void* d_out, int out_size, void* d_ws, size_t ws_size,
                              hipStream_t stream) {
  const float* feature = (const float*)d_in[0];
  const float* temp    = (const float*)d_in[1];
  const float* spat    = (const float*)d_in[2];
  const float* prev    = (const float*)d_in[3];
  const float* bnf_g = (const float*)d_in[4],  *bnf_b = (const float*)d_in[5];
  const float* bnf_m = (const float*)d_in[6],  *bnf_v = (const float*)d_in[7];
  const float* bnh_g = (const float*)d_in[8],  *bnh_b = (const float*)d_in[9];
  const float* bnh_m = (const float*)d_in[10], *bnh_v = (const float*)d_in[11];
  const float* bn1_g = (const float*)d_in[12], *bn1_b = (const float*)d_in[13];
  const float* bn1_m = (const float*)d_in[14], *bn1_v = (const float*)d_in[15];
  const float* bn2_g = (const float*)d_in[16], *bn2_b = (const float*)d_in[17];
  const float* bn2_m = (const float*)d_in[18], *bn2_v = (const float*)d_in[19];
  const float* Wt = (const float*)d_in[20], *bt = (const float*)d_in[21];
  const float* Ws = (const float*)d_in[22], *bs = (const float*)d_in[23];
  const float* Wf = (const float*)d_in[24], *Wh2 = (const float*)d_in[25];
  const float* gru_wih = (const float*)d_in[26];
  const float* gru_whh = (const float*)d_in[27];
  const float* gru_b   = (const float*)d_in[28];
  const float* Wp = (const float*)d_in[29], *bp = (const float*)d_in[30];
  const float* Wg = (const float*)d_in[31], *bg = (const float*)d_in[32];

  float* out  = (float*)d_out;
  float* pout = out;                                  // (P,64)
  float* gout = out + (size_t)PP * NPOUT;             // (P,32)
  float* hout = out + (size_t)PP * (NPOUT + NGOUT);   // (P,256)

  // workspace: f32 region, then bf16 region (all 16B aligned)
  float* attnT = (float*)d_ws;                        // (E,P)
  float* smax  = attnT + (size_t)EE * PP;             // 32
  float* Hagg  = smax + 32;                           // (P,H)
  __bf16* catB = (__bf16*)(Hagg + (size_t)PP * HH);   // (P,512) bf16
  __bf16* WtB  = catB + (size_t)PP * 512;             // (128,256)
  __bf16* WsTB = WtB + AA * HH;                       // (256,128)
  __bf16* WfB  = WsTB + HH * AA;                      // (256,1024)
  __bf16* Wh2B = WfB + HH * DD;                       // (256,512)
  __bf16* wihB = Wh2B + HH * 512;                     // (3,256,512)
  __bf16* whhB = wihB + 3 * HH * 512;                 // (3,256,256)
  __bf16* WpB  = whhB + 3 * HH * HH;                  // (64,256)
  __bf16* WgB  = WpB + NPOUT * HH;                    // (32,256)

  // weight prep (bf16, transposed where needed)
  k_cvt<<<(AA * HH + 255) / 256, 256, 0, stream>>>(Wt, WtB, AA * HH);
  k_transcvt<<<(AA * HH) / 256, 256, 0, stream>>>(Ws, WsTB);
  k_cvt<<<(HH * DD + 255) / 256, 256, 0, stream>>>(Wf, WfB, HH * DD);
  k_cvt<<<(HH * 512 + 255) / 256, 256, 0, stream>>>(Wh2, Wh2B, HH * 512);
  k_cvt<<<(3 * HH * 512 + 255) / 256, 256, 0, stream>>>(gru_wih, wihB, 3 * HH * 512);
  k_cvt<<<(3 * HH * HH + 255) / 256, 256, 0, stream>>>(gru_whh, whhB, 3 * HH * HH);
  k_cvt<<<(NPOUT * HH + 255) / 256, 256, 0, stream>>>(Wp, WpB, NPOUT * HH);
  k_cvt<<<(NGOUT * HH + 255) / 256, 256, 0, stream>>>(Wg, WgB, NGOUT * HH);

  k_attn<<<PP / 16, 256, 0, stream>>>(temp, spat, bnh_g, bnh_b, bnh_m, bnh_v,
                                      WtB, bt, WsTB, bs, attnT);
  k_softmax<<<EE, 256, 0, stream>>>(attnT, smax);
  k_hagg<<<PP, 256, 0, stream>>>(spat, bnh_g, bnh_b, bnh_m, bnh_v, attnT, smax, Hagg);
  k_feat<<<PP / 16, 256, 0, stream>>>(feature, bnf_g, bnf_b, bnf_m, bnf_v, WfB, catB);
  k_he<<<PP / 16, 256, 0, stream>>>(temp, Hagg, bn2_g, bn2_b, bn2_m, bn2_v, Wh2B, catB);
  k_gru<<<PP / 16, 256, 0, stream>>>(prev, catB, wihB, whhB, gru_b, hout);
  k_heads<<<PP / 16, 96, 0, stream>>>(hout, bn1_g, bn1_b, bn1_m, bn1_v,
                                      WpB, bp, WgB, bg, pout, gout);
}
